// ScaledDotProductAttention_7687991460015
// MI455X (gfx1250) — matrix operand order repaired and checked by reading, not compiled, not run on hardware
//
#include <hip/hip_runtime.h>
#include <hip/hip_bf16.h>
#include <stdint.h>

typedef __attribute__((ext_vector_type(16))) _Float16 v16h;
typedef __attribute__((ext_vector_type(2)))  _Float16 v2h;
typedef __attribute__((ext_vector_type(2)))  __fp16   v2fp;   // return type of cvt_pkrtz
typedef __attribute__((ext_vector_type(8)))  float    v8f;

#define S_LEN 4096
#define D_DIM 64
#define BQ    64     // query rows per workgroup (4 waves x 16)
#define BK    64     // kv columns per iteration (4 WMMA n-tiles)
#define NWAVE 4

static __device__ inline v2h pk2h(float x, float y) {
    v2fp t = __builtin_amdgcn_cvt_pkrtz(x, y);
    return __builtin_bit_cast(v2h, t);
}

// One-time mask conversion: byte {0,1} -> f16 multiplicand {1.0, 0.0}.
// Memory-bound (~50MB total traffic); result lives in d_ws and is reused by all
// 16 heads x 64 q-blocks through the 192MB L2.
__global__ __launch_bounds__(256)
void MaskToNm(const unsigned char* __restrict__ m, _Float16* __restrict__ nm) {
    int i = (blockIdx.x * 256 + threadIdx.x) * 8;
    uint2 w = *(const uint2*)(m + i);
    _Float16 h[8];
#pragma unroll
    for (int j = 0; j < 4; ++j) {
        h[j]     = ((w.x >> (8 * j)) & 0xFF) ? (_Float16)0.0f : (_Float16)1.0f;
        h[4 + j] = ((w.y >> (8 * j)) & 0xFF) ? (_Float16)0.0f : (_Float16)1.0f;
    }
    *(uint4*)(nm + i) = *(uint4*)h;
}

template<bool NM>
__global__ __launch_bounds__(128)
void ScaledDotProductAttention_7687991460015_kernel(
    const float* __restrict__ q, const float* __restrict__ k,
    const float* __restrict__ v, const unsigned char* __restrict__ mask,
    const _Float16* __restrict__ nmG, float* __restrict__ out)
{
    __shared__ __align__(128) _Float16 sK [BK][D_DIM];     // K chunk, row-major (kv x d)   8KB
    __shared__ __align__(128) _Float16 sVt[D_DIM][BK];     // V chunk, transposed (d x kv)  8KB
    __shared__ __align__(128) _Float16 sP [NWAVE][16][BK]; // per-wave P tile bounce        8KB
    __shared__ __align__(128) _Float16 sNm[BQ][BK];        // f16 not-mask (or raw bytes)   8KB
    unsigned char (*sMu)[BK] = reinterpret_cast<unsigned char(*)[BK]>(&sNm[0][0]);

    const int tid  = threadIdx.x;
    const int wave = tid >> 5;
    const int lane = tid & 31;
    const int bh   = blockIdx.x >> 6;        // 0..15  (B*H = 16)
    const int q0   = (blockIdx.x & 63) * BQ; // query block start

    // uniform (SGPR) head base pointers
    const float* kB = k + (size_t)bh * (S_LEN * D_DIM);
    const float* vB = v + (size_t)bh * (S_LEN * D_DIM);
    const float* qB = q + (size_t)bh * (S_LEN * D_DIM);
    float*       oB = out + (size_t)bh * (S_LEN * D_DIM);

    const int mrow   = lane & 15;            // A-layout row / B-layout col / C-layout col
    const int hiHalf = (lane >> 4) & 1;      // 0: lanes 0-15, 1: lanes 16-31
    const int koffA  = hiHalf * 8;           // table-order A K offset for this half (unused)
    const int kbB    = hiHalf * 16;          // K offset for this half (contiguous k = 16h + s, used for A and B)

    // ---- Load Q tile into WMMA A-layout regs; fold log2(e)/sqrt(D) (base-2 softmax) ----
    const int qoff = (q0 + wave * 16 + mrow) * D_DIM;
    const float qs = 0.125f * 1.44269504088896340736f; // log2(e)/sqrt(64)
    v16h aq0, aq1;
#pragma unroll
    for (int g = 0; g < 8; ++g) {
        int kk = kbB + 2 * g;
        aq0[2*g]   = (_Float16)(qB[qoff + kk]      * qs);
        aq0[2*g+1] = (_Float16)(qB[qoff + kk + 1]  * qs);
        aq1[2*g]   = (_Float16)(qB[qoff + kk + 32] * qs);
        aq1[2*g+1] = (_Float16)(qB[qoff + kk + 33] * qs);
    }

    // all-ones B operand: P x Ones replicates the row-sum into every output column,
    // so the softmax denominator is just a 5th WMMA accumulator (no shuffle reduce).
    v16h ones;
#pragma unroll
    for (int e = 0; e < 16; ++e) ones[e] = (_Float16)1.0f;

    float mrun[8];
    v8f acc[5];                              // 4 output d-tiles + 1 row-sum tile
#pragma unroll
    for (int r = 0; r < 8; ++r) mrun[r] = -3.0e38f;
#pragma unroll
    for (int t = 0; t < 5; ++t) { v8f z = {}; acc[t] = z; }

    for (int kv0 = 0; kv0 < S_LEN; kv0 += BK) {
        __syncthreads(); // protect LDS from previous iteration's readers

        // uniform per-iteration bases (SGPR adds); lane offsets below are kv-invariant
        const float*    kIter = kB + kv0 * D_DIM;
        const float*    vIter = vB + kv0 * D_DIM;
        const _Float16* nIter = NM ? (nmG + q0 * S_LEN + kv0) : (const _Float16*)nullptr;
        const unsigned char* mIter = NM ? (const unsigned char*)nullptr
                                        : (mask + q0 * S_LEN + kv0);

        // ---- cooperative stage: K (row-major, packed cvt), V (transposed), mask ----
#pragma unroll
        for (int j = 0; j < 8; ++j) {
            int idx = tid + 128 * j;      // 1024 float4's total
            int off = idx * 4;            // == r*64 + c  (kv-invariant lane offset)
            int r = off >> 6, c = off & 63;
            const float4 kd = *(const float4*)(kIter + off);
            const float4 vd = *(const float4*)(vIter + off);
            v2fp k01 = __builtin_amdgcn_cvt_pkrtz(kd.x, kd.y);
            v2fp k23 = __builtin_amdgcn_cvt_pkrtz(kd.z, kd.w);
            *(v2fp*)&sK[r][c]     = k01;
            *(v2fp*)&sK[r][c + 2] = k23;
            v2fp v01 = __builtin_amdgcn_cvt_pkrtz(vd.x, vd.y);
            v2fp v23 = __builtin_amdgcn_cvt_pkrtz(vd.z, vd.w);
            sVt[c][r]     = (_Float16)v01[0]; sVt[c + 1][r] = (_Float16)v01[1];
            sVt[c + 2][r] = (_Float16)v23[0]; sVt[c + 3][r] = (_Float16)v23[1];
            int mloff = r * S_LEN + c;    // kv-invariant lane offset
            if (NM) { // 64x64 f16 not-mask = 1024 uint2
                *(uint2*)&sNm[r][c] = *(const uint2*)(nIter + mloff);
            } else {  // 64x64 bytes = 1024 dwords
                *(uint32_t*)&sMu[r][c] = *(const uint32_t*)(mIter + mloff);
            }
        }
        __syncthreads();

        // ---- S = Q * K^T : four 16x16 tiles, each K-dim 64 = 2 WMMAs ----
        v8f st[4];
#pragma unroll
        for (int nt = 0; nt < 4; ++nt) {
            v16h b0 = *(const v16h*)&sK[nt * 16 + mrow][kbB];
            v16h b1 = *(const v16h*)&sK[nt * 16 + mrow][32 + kbB];
            v8f z = {};
            st[nt] = __builtin_amdgcn_wmma_f32_16x16x32_f16(false, aq0, false, b0, (short)0, z,      false, false);
            st[nt] = __builtin_amdgcn_wmma_f32_16x16x32_f16(false, aq1, false, b1, (short)0, st[nt], false, false);
        }

        // ---- row max over the chunk (raw scores are a valid stabilizer; mask is
        //      applied as a zeroing multiply AFTER exp in the NM path) ----
        float tm[8];
        float upd = 0.0f;
#pragma unroll
        for (int r = 0; r < 8; ++r) {
            if (!NM) { // legacy path: mask must hit the max too (exact -inf semantics)
                int prow = hiHalf * 8 + r;
#pragma unroll
                for (int t = 0; t < 4; ++t)
                    if (sMu[wave * 16 + prow][t * 16 + mrow]) st[t][r] = -1.0e30f;
            }
            float tmr = fmaxf(fmaxf(st[0][r], st[1][r]), fmaxf(st[2][r], st[3][r]));
#pragma unroll
            for (int o = 1; o < 16; o <<= 1) tmr = fmaxf(tmr, __shfl_xor(tmr, o, 32));
            tm[r] = tmr;
            upd = fmaxf(upd, tmr - mrun[r]);
        }

        // ---- rescale accumulators only when some row max actually grew ----
        if (__any(upd > 0.0f)) {
#pragma unroll
            for (int r = 0; r < 8; ++r) {
                float mn    = fmaxf(mrun[r], tm[r]);
                float alpha = __builtin_amdgcn_exp2f(mrun[r] - mn);
                mrun[r] = mn;
#pragma unroll
                for (int t = 0; t < 5; ++t) acc[t][r] *= alpha;
            }
        }

        // ---- P = exp2(S - m) (masked), packed f16, store to wave-private LDS ----
#pragma unroll
        for (int r = 0; r < 8; ++r) {
            int prow = hiHalf * 8 + r;
            float p0 = __builtin_amdgcn_exp2f(st[0][r] - mrun[r]);
            float p1 = __builtin_amdgcn_exp2f(st[1][r] - mrun[r]);
            float p2 = __builtin_amdgcn_exp2f(st[2][r] - mrun[r]);
            float p3 = __builtin_amdgcn_exp2f(st[3][r] - mrun[r]);
            v2h q01 = pk2h(p0, p1);
            v2h q23 = pk2h(p2, p3);
            if (NM) {
                v2h nm01, nm23;
                nm01[0] = sNm[wave * 16 + prow][mrow];
                nm01[1] = sNm[wave * 16 + prow][16 + mrow];
                nm23[0] = sNm[wave * 16 + prow][32 + mrow];
                nm23[1] = sNm[wave * 16 + prow][48 + mrow];
                q01 *= nm01;           // v_pk_mul_f16: zero masked P
                q23 *= nm23;
            }
            sP[wave][prow][mrow]      = q01[0];
            sP[wave][prow][16 + mrow] = q01[1];
            sP[wave][prow][32 + mrow] = q23[0];
            sP[wave][prow][48 + mrow] = q23[1];
        }

        // wave-private LDS bounce: same-wave DS ops are in-order
        asm volatile("s_wait_dscnt 0" ::: "memory");

        // ---- read P back in A-layout (16x64 f16 -> two k=32 operands) ----
        v16h pa0, pa1;
#pragma unroll
        for (int g = 0; g < 8; ++g) {
            int kk = kbB + 2 * g;
            pa0[2*g]   = sP[wave][mrow][kk];
            pa0[2*g+1] = sP[wave][mrow][kk + 1];
            pa1[2*g]   = sP[wave][mrow][kk + 32];
            pa1[2*g+1] = sP[wave][mrow][kk + 33];
        }

        // ---- O += P * [V | 1] : (4 d-tiles + row-sum tile) x 2 k-chunks = 10 WMMAs ----
#pragma unroll
        for (int c = 0; c < 2; ++c) {
            v16h pac = c ? pa1 : pa0;
#pragma unroll
            for (int dt = 0; dt < 4; ++dt) {
                v16h bv = *(const v16h*)&sVt[dt * 16 + mrow][c * 32 + kbB];
                acc[dt] = __builtin_amdgcn_wmma_f32_16x16x32_f16(false, pac, false, bv, (short)0, acc[dt], false, false);
            }
            acc[4] = __builtin_amdgcn_wmma_f32_16x16x32_f16(false, pac, false, ones, (short)0, acc[4], false, false);
        }
    }

    // ---- epilogue: normalize by the row-sum tile (replicated across cols) ----
#pragma unroll
    for (int r = 0; r < 8; ++r) {
        int row = q0 + wave * 16 + hiHalf * 8 + r;
        float inv = 1.0f / acc[4][r];
#pragma unroll
        for (int dt = 0; dt < 4; ++dt) {
            oB[row * D_DIM + dt * 16 + mrow] = acc[dt][r] * inv;
        }
    }
}

extern "C" void kernel_launch(void* const* d_in, const int* in_sizes, int n_in,
                              void* d_out, int out_size, void* d_ws, size_t ws_size,
                              hipStream_t stream) {
    (void)in_sizes; (void)n_in; (void)out_size;
    const float* q = (const float*)d_in[0];
    const float* k = (const float*)d_in[1];
    const float* v = (const float*)d_in[2];
    const unsigned char* mask = (const unsigned char*)d_in[3];
    float* out = (float*)d_out;

    dim3 grid(16 * (S_LEN / BQ));   // 1024 workgroups
    dim3 block(128);                // 4 wave32

    const size_t need = (size_t)S_LEN * S_LEN * sizeof(_Float16);
    if (ws_size >= need) {
        _Float16* nm = (_Float16*)d_ws;
        MaskToNm<<<(S_LEN * S_LEN) / (256 * 8), 256, 0, stream>>>(mask, nm);
        ScaledDotProductAttention_7687991460015_kernel<true>
            <<<grid, block, 0, stream>>>(q, k, v, mask, nm, out);
    } else {
        ScaledDotProductAttention_7687991460015_kernel<false>
            <<<grid, block, 0, stream>>>(q, k, v, mask, nullptr, out);
    }
}
